// multi_head_attention_24086176596635
// MI455X (gfx1250) — matrix operand order repaired and checked by reading, not compiled, not run on hardware
//
#include <hip/hip_runtime.h>
#include <hip/hip_bf16.h>
#include <math.h>

typedef __attribute__((ext_vector_type(16))) _Float16 v16h;
typedef __attribute__((ext_vector_type(8)))  _Float16 v8h;
typedef __attribute__((ext_vector_type(8)))  float    v8f;

#define B_  4
#define N_  2048
#define IN_ 16
#define E_  512
#define H_  8
#define DH_ 64

// gfx1250 async global->LDS path via inline asm (ASYNCcnt-tracked).
#if defined(__gfx1250__)
#define USE_ASYNC_LDS 1
#else
#define USE_ASYNC_LDS 0
#endif

// ---------------------------------------------------------------------------
// 0) Weight prep: Wt[n][k] = (f16) W[k][n].  Makes every GEMM B-fragment a
//    single contiguous 32 B load.
// ---------------------------------------------------------------------------
__global__ void transpose_w_kernel(const float* __restrict__ W,
                                   _Float16* __restrict__ Wt) {
  __shared__ float tile[16][17];
  const int tx = threadIdx.x & 15, ty = threadIdx.x >> 4;   // 256 thr = 16x16
  const int kb = blockIdx.x * 16, nb = blockIdx.y * 16;
  tile[ty][tx] = W[(size_t)(kb + ty) * E_ + nb + tx];
  __syncthreads();
  Wt[(size_t)(nb + ty) * E_ + kb + tx] = (_Float16)tile[tx][ty];
}

// ---------------------------------------------------------------------------
// 1) Embedding MLP: e = relu(relu(x@W1+b1)@W2+b2) -> f16.  Tiny work.
// ---------------------------------------------------------------------------
__global__ void mlp_embed_kernel(const float* __restrict__ x,
                                 const float* __restrict__ W1,
                                 const float* __restrict__ b1,
                                 const float* __restrict__ W2,
                                 const float* __restrict__ b2,
                                 _Float16* __restrict__ e_h) {
  __shared__ float xr[IN_];
  __shared__ float hs[128];
  const int row = blockIdx.x;      // 0 .. B*N-1
  const int t   = threadIdx.x;     // 0 .. 127
  if (t < IN_) xr[t] = x[(size_t)row * IN_ + t];
  __syncthreads();
  float h = b1[t];
#pragma unroll
  for (int k = 0; k < IN_; ++k) h += xr[k] * W1[k * 128 + t];
  hs[t] = fmaxf(h, 0.f);
  __syncthreads();
#pragma unroll
  for (int cc = 0; cc < 4; ++cc) {
    const int c = t + 128 * cc;
    float acc = b2[c];
    for (int k = 0; k < 128; ++k) acc += hs[k] * W2[k * E_ + c];
    e_h[(size_t)row * E_ + c] = (_Float16)fmaxf(acc, 0.f);
  }
}

// ---------------------------------------------------------------------------
// WMMA GEMM: out[M,N] = A[M,K](f16, row-major) @ Wt[N,K](f16)^T + bias.
// 32x32 output tile per wave: 2 A-frags + 2 B-frags -> 4 WMMA per K-step.
// MODE: 0 = f16 row-major out, 1 = f32 row-major out,
//       2 = f16 out transposed per column into vT[b][ncol][nseq] (for V).
// ---------------------------------------------------------------------------
template <int MODE>
__global__ void wmma_gemm_kernel(const _Float16* __restrict__ A,
                                 const _Float16* __restrict__ Wt,
                                 const float* __restrict__ bias,
                                 void* __restrict__ out,
                                 int M, int N, int K) {
  const int lane = threadIdx.x & 31;
  const int wave = threadIdx.x >> 5;
  const int tile = blockIdx.x * 8 + wave;
  const int ntiles = N >> 5;                 // 32-wide col tiles
  const int mt = tile / ntiles;
  const int nt = tile - mt * ntiles;
  const int li = lane & 15;
  const int sel = lane >> 4;

  const _Float16* arow0 = A + (size_t)(mt * 32 + li) * K;
  const _Float16* arow1 = A + (size_t)(mt * 32 + 16 + li) * K;
  const _Float16* wrow0 = Wt + (size_t)(nt * 32 + li) * K;
  const _Float16* wrow1 = Wt + (size_t)(nt * 32 + 16 + li) * K;

  v8f acc[2][2] = {};
  for (int k0 = 0; k0 < K; k0 += 32) {
    __builtin_prefetch(wrow0 + k0 + 128, 0, 0);   // global_prefetch_b8
    v16h a[2], b[2];
    {
      v8h lo0 = *(const v8h*)(arow0 + k0 + sel * 16);
      v8h hi0 = *(const v8h*)(arow0 + k0 + 8 + sel * 16);
      v8h lo1 = *(const v8h*)(arow1 + k0 + sel * 16);
      v8h hi1 = *(const v8h*)(arow1 + k0 + 8 + sel * 16);
#pragma unroll
      for (int i = 0; i < 8; ++i) {
        a[0][i] = lo0[i]; a[0][8 + i] = hi0[i];
        a[1][i] = lo1[i]; a[1][8 + i] = hi1[i];
      }
    }
    b[0] = *(const v16h*)(wrow0 + k0 + sel * 16);
    b[1] = *(const v16h*)(wrow1 + k0 + sel * 16);
#pragma unroll
    for (int am = 0; am < 2; ++am)
#pragma unroll
      for (int bn = 0; bn < 2; ++bn)
        acc[am][bn] = __builtin_amdgcn_wmma_f32_16x16x32_f16(
            false, a[am], false, b[bn], (short)0, acc[am][bn], false, false);
  }

#pragma unroll
  for (int am = 0; am < 2; ++am) {
#pragma unroll
    for (int bn = 0; bn < 2; ++bn) {
      const int ncol = nt * 32 + bn * 16 + li;
      const float bv = bias[ncol];
      if (MODE == 2) {
        // vT[b][ncol][nseq]: lane writes 8 contiguous f16 (one b128 store)
        const int m0 = mt * 32 + am * 16 + 8 * sel;
        const int bb = m0 >> 11;               // N_ = 2048 rows per batch
        const int ns0 = m0 & (N_ - 1);
        v8h pk;
#pragma unroll
        for (int r = 0; r < 8; ++r) pk[r] = (_Float16)(acc[am][bn][r] + bv);
        *(v8h*)((_Float16*)out + ((size_t)bb * E_ + ncol) * N_ + ns0) = pk;
      } else {
#pragma unroll
        for (int r = 0; r < 8; ++r) {
          const int m = mt * 32 + am * 16 + r + 8 * sel;
          const float v = acc[am][bn][r] + bv;
          if (MODE == 0) ((_Float16*)out)[(size_t)m * N + ncol] = (_Float16)v;
          else           ((float*)out)[(size_t)m * N + ncol] = v;
        }
      }
    }
  }
}

// ---------------------------------------------------------------------------
// Attention: one block (4 waves, 128 thr) per (b, h, 16-row q-tile).
// LDS: 16x2048 f32 scores (128 KB) + 16x2048 f16 probs (64 KB) + 2 KB q tile.
// Probs hit HBM exactly once (the 536 MB roofline term).
// ---------------------------------------------------------------------------
__global__ void attn_kernel(const _Float16* __restrict__ q,
                            const _Float16* __restrict__ k,
                            const _Float16* __restrict__ vT,
                            const float* __restrict__ mask,
                            float* __restrict__ probs,
                            _Float16* __restrict__ ctx) {
  extern __shared__ float S[];                        // [16][N_] f32
  _Float16* P16 = (_Float16*)(S + 16 * N_);           // [16][N_] f16
  __shared__ _Float16 qtile[16 * DH_];                // staged q tile (2 KB)
  __shared__ float red[16][8];
  __shared__ float rowstat[16];

  const int tid  = threadIdx.x;
  const int lane = tid & 31;
  const int wave = tid >> 5;
  const int blk  = blockIdx.x;
  const int qt = blk & 127;                           // N_/16 q tiles
  const int bh = blk >> 7;
  const int b  = bh >> 3;
  const int h  = bh & 7;
  const int li = lane & 15;
  const int sel = lane >> 4;

  const size_t headoff = (size_t)h * DH_;
  const _Float16* qbase = q + (size_t)b * N_ * E_ + headoff;
  const _Float16* kbase = k + (size_t)b * N_ * E_ + headoff;

  // ---- Stage q tile into LDS (async global->LDS on gfx1250) ----
  {
    const int r  = tid >> 3;                // 16 rows
    const int c8 = (tid & 7) * 8;           // 8 x 16B chunks per row
    const _Float16* gsrc = qbase + (size_t)(qt * 16 + r) * E_ + c8;
    _Float16* ldst = qtile + r * DH_ + c8;
#if USE_ASYNC_LDS
    // Flat LDS-aperture addresses carry the LDS byte offset in addr[31:0]
    // (ISA 10.2), so truncating the generic pointer yields the VDST operand.
    const unsigned lds_addr = (unsigned)(unsigned long long)ldst;
    const unsigned long long gaddr = (unsigned long long)gsrc;
    asm volatile("global_load_async_to_lds_b128 %0, %1, off"
                 :: "v"(lds_addr), "v"(gaddr)
                 : "memory");
    asm volatile("s_wait_asynccnt 0" ::: "memory");
#else
    *(v8h*)ldst = *(const v8h*)gsrc;
#endif
  }
  __syncthreads();
  const _Float16* qrow = qtile + (size_t)li * DH_;    // ds_load fragments

  // ---- Phase 1: raw scores -> LDS (WMMA) ----
  v16h aq[2];
#pragma unroll
  for (int c = 0; c < 2; ++c) {
    v8h lo = *(const v8h*)(qrow + c * 32 + sel * 16);
    v8h hi = *(const v8h*)(qrow + c * 32 + 8 + sel * 16);
#pragma unroll
    for (int i = 0; i < 8; ++i) { aq[c][i] = lo[i]; aq[c][8 + i] = hi[i]; }
  }
  for (int kt = wave; kt < N_ / 16; kt += 4) {
    v8f acc = {};
    const _Float16* krow = kbase + (size_t)(kt * 16 + li) * E_;
#pragma unroll
    for (int c = 0; c < 2; ++c) {
      v16h bfr = *(const v16h*)(krow + c * 32 + sel * 16);
      acc = __builtin_amdgcn_wmma_f32_16x16x32_f16(false, aq[c], false, bfr,
                                                   (short)0, acc, false, false);
    }
#pragma unroll
    for (int r = 0; r < 8; ++r)
      S[(r + 8 * sel) * N_ + kt * 16 + li] = acc[r];
  }
  __syncthreads();

  // ---- Phase 2: scale + mask (coalesced) ----
  const float scale = 0.125f;                         // 1/sqrt(DH)
  const float* mrow = mask + ((size_t)b * N_ + qt * 16) * N_;
  for (int t = tid; t < 16 * N_; t += 128) {
    const int i = t >> 11;
    const int j = t & (N_ - 1);
    const float s = S[t] * scale;
    S[t] = (mrow[(size_t)i * N_ + j] < 0.1f) ? -1e10f : s;
  }
  __syncthreads();

  // ---- Phase 3: row softmax (8 threads per row) ----
  {
    const int row = tid >> 3, part = tid & 7;
    float m = -INFINITY;
    for (int j = part * 256; j < part * 256 + 256; ++j)
      m = fmaxf(m, S[row * N_ + j]);
    red[row][part] = m;
    __syncthreads();
    if (part == 0) {
      float mm = red[row][0];
      for (int p = 1; p < 8; ++p) mm = fmaxf(mm, red[row][p]);
      rowstat[row] = mm;
    }
    __syncthreads();
    const float rm = rowstat[row];
    float sum = 0.f;
    for (int j = part * 256; j < part * 256 + 256; ++j) {
      const float p = __expf(S[row * N_ + j] - rm);
      S[row * N_ + j] = p;
      sum += p;
    }
    red[row][part] = sum;
    __syncthreads();
    if (part == 0) {
      float ss = 0.f;
      for (int p = 0; p < 8; ++p) ss += red[row][p];
      rowstat[row] = 1.f / ss;
    }
    __syncthreads();
  }

  // normalize: one coalesced f32 store to HBM + f16 copy into LDS
  float* prow = probs + ((size_t)bh * N_ + qt * 16) * N_;
  for (int t = tid; t < 16 * N_; t += 128) {
    const int i = t >> 11;
    const int j = t & (N_ - 1);
    const float p = S[t] * rowstat[i];
    prow[(size_t)i * N_ + j] = p;
    P16[t] = (_Float16)p;
  }
  __syncthreads();

  // ---- Phase 4: ctx = P @ V ; wave w owns head dims [w*16, w*16+16) ----
  v8f cacc = {};
  const int nd = wave * 16 + li;
  const _Float16* vrow = vT + ((size_t)b * E_ + headoff + nd) * N_;
  const _Float16* psrc = P16 + (size_t)li * N_;
  for (int k0 = 0; k0 < N_; k0 += 32) {
    v16h a;
    v8h lo = *(const v8h*)(psrc + k0 + sel * 16);      // ds_load_b128
    v8h hi = *(const v8h*)(psrc + k0 + 8 + sel * 16);
#pragma unroll
    for (int i = 0; i < 8; ++i) { a[i] = lo[i]; a[8 + i] = hi[i]; }
    v16h bfr = *(const v16h*)(vrow + k0 + sel * 16);   // contiguous 32 B
    cacc = __builtin_amdgcn_wmma_f32_16x16x32_f16(false, a, false, bfr,
                                                  (short)0, cacc, false, false);
  }
#pragma unroll
  for (int r = 0; r < 8; ++r) {
    const int m = qt * 16 + r + 8 * sel;
    ctx[((size_t)b * N_ + m) * E_ + headoff + nd] = (_Float16)cacc[r];
  }
}

// ---------------------------------------------------------------------------
extern "C" void kernel_launch(void* const* d_in, const int* in_sizes, int n_in,
                              void* d_out, int out_size, void* d_ws, size_t ws_size,
                              hipStream_t stream) {
  (void)in_sizes; (void)n_in; (void)out_size; (void)ws_size;
  const float* x    = (const float*)d_in[0];
  const float* mask = (const float*)d_in[1];
  const float* W1   = (const float*)d_in[2];
  const float* b1   = (const float*)d_in[3];
  const float* W2   = (const float*)d_in[4];
  const float* b2   = (const float*)d_in[5];
  const float* Wq   = (const float*)d_in[6];
  const float* bq   = (const float*)d_in[7];
  const float* Wk   = (const float*)d_in[8];
  const float* bk   = (const float*)d_in[9];
  const float* Wv   = (const float*)d_in[10];
  const float* bv   = (const float*)d_in[11];
  const float* Wo   = (const float*)d_in[12];
  const float* bo   = (const float*)d_in[13];

  float* out   = (float*)d_out;                         // [B,N,E]
  float* probs = out + (size_t)B_ * N_ * E_;            // [B,H,N,N]

  char* ws = (char*)d_ws;
  const size_t mat_h = (size_t)B_ * N_ * E_ * sizeof(_Float16);   // 8 MB
  const size_t w_h   = (size_t)E_ * E_ * sizeof(_Float16);        // 0.5 MB
  _Float16* e_h   = (_Float16*)(ws);
  _Float16* q_h   = (_Float16*)(ws + 1 * mat_h);
  _Float16* k_h   = (_Float16*)(ws + 2 * mat_h);
  _Float16* vT_h  = (_Float16*)(ws + 3 * mat_h);        // [B][E][N]
  _Float16* ctx_h = (_Float16*)(ws + 4 * mat_h);
  _Float16* Wt_q  = (_Float16*)(ws + 5 * mat_h);
  _Float16* Wt_k  = (_Float16*)(ws + 5 * mat_h + 1 * w_h);
  _Float16* Wt_v  = (_Float16*)(ws + 5 * mat_h + 2 * w_h);
  _Float16* Wt_o  = (_Float16*)(ws + 5 * mat_h + 3 * w_h);

  const int M = B_ * N_;                                // 8192

  // 0) transpose + f16-convert the four E x E weights
  dim3 tgrid(E_ / 16, E_ / 16);
  transpose_w_kernel<<<tgrid, 256, 0, stream>>>(Wq, Wt_q);
  transpose_w_kernel<<<tgrid, 256, 0, stream>>>(Wk, Wt_k);
  transpose_w_kernel<<<tgrid, 256, 0, stream>>>(Wv, Wt_v);
  transpose_w_kernel<<<tgrid, 256, 0, stream>>>(Wo, Wt_o);

  // 1) embedding MLP -> f16 e
  mlp_embed_kernel<<<M, 128, 0, stream>>>(x, W1, b1, W2, b2, e_h);

  // 2) Q/K/V projections. 32x32 tiles: (8192/32)*(512/32)=4096 waves = 512 blk
  const int gemm_blocks = (M / 32) * (E_ / 32) / 8;
  wmma_gemm_kernel<0><<<gemm_blocks, 256, 0, stream>>>(e_h, Wt_q, bq, q_h, M, E_, E_);
  wmma_gemm_kernel<0><<<gemm_blocks, 256, 0, stream>>>(e_h, Wt_k, bk, k_h, M, E_, E_);
  wmma_gemm_kernel<2><<<gemm_blocks, 256, 0, stream>>>(e_h, Wt_v, bv, vT_h, M, E_, E_);

  // 3) attention (192 KB dynamic LDS per workgroup)
  const size_t smem = (size_t)16 * N_ * (sizeof(float) + sizeof(_Float16));
  attn_kernel<<<B_ * H_ * (N_ / 16), 128, smem, stream>>>(q_h, k_h, vT_h, mask,
                                                          probs, ctx_h);

  // 4) output projection -> f32 straight into d_out
  wmma_gemm_kernel<1><<<gemm_blocks, 256, 0, stream>>>(ctx_h, Wt_o, bo, out, M, E_, E_);
}
